// TtCrossAttention_44959717655250
// MI455X (gfx1250) — matrix operand order repaired and checked by reading, not compiled, not run on hardware
//
#include <hip/hip_runtime.h>
#include <hip/hip_bf16.h>

typedef __bf16 bf16_t;
typedef __attribute__((ext_vector_type(16))) __bf16 v16bf;
typedef __attribute__((ext_vector_type(8)))  __bf16 v8bf;
typedef __attribute__((ext_vector_type(8)))  float  v8f;

#define HEADS     8
#define DIM_HEAD  64
#define QUERY_DIM 512
#define CROSS_DIM 768
#define INNER     512
#define BATCH     2
#define SQ        4096
#define SKV       1024

// fp32 -> bf16, round-to-nearest-even (convert pass, P tile and epilogues)
__device__ __forceinline__ bf16_t f2bf(float f) {
  union { float f; unsigned u; } v; v.f = f;
  unsigned r = v.u + 0x7FFFu + ((v.u >> 16) & 1u);
  unsigned short h = (unsigned short)(r >> 16);
  return __builtin_bit_cast(bf16_t, h);
}

__device__ __forceinline__ v8f wmma_bf16(v16bf a, v16bf b, v8f c) {
  // D = A(16x32 bf16) * B(32x16 bf16) + C(16x16 f32)
  return __builtin_amdgcn_wmma_f32_16x16x32_bf16(false, a, false, b, (short)0, c,
                                                 false, false);
}

// ---- Fragment loaders (k = 16*half + s in both A and B; not the ISA table order) ----
// A 16x32 (16-bit): lane row = lane&15, hi half selects K sub-range.
// Lane holds K = {half*16..+15}: two contiguous 16-byte chunks.
__device__ __forceinline__ v16bf load_a16(const bf16_t* base, int ld, int row,
                                          int half, int k0) {
  const bf16_t* p0 = base + (long)row * ld + k0 + half * 16;
  v8bf lo = *(const v8bf*)p0;
  v8bf hi = *(const v8bf*)(p0 + 8);
  return __builtin_shufflevector(lo, hi, 0, 1, 2, 3, 4, 5, 6, 7,
                                 8, 9, 10, 11, 12, 13, 14, 15);
}

// B 32x16 (16-bit): lane = column n, hi selects 16 contiguous K values.
// Source must have B-columns contiguous in K (row-major [N][K]); 32B aligned.
__device__ __forceinline__ v16bf load_b16(const bf16_t* base, int ld, int col,
                                          int kgrp, int k0) {
  return *(const v16bf*)(base + (long)col * ld + k0 + kgrp * 16);
}

// ---------------------------------------------------------------------------
// One-shot fp32 -> bf16 conversion (memory bound): 8 elements per thread.
// ---------------------------------------------------------------------------
__global__ __launch_bounds__(256)
void cvt_bf16_kernel(const float* __restrict__ in, bf16_t* __restrict__ out,
                     long n) {
  long i = ((long)blockIdx.x * blockDim.x + threadIdx.x) * 8;
  if (i >= n) return;
  float4 x0 = *(const float4*)(in + i);
  float4 x1 = *(const float4*)(in + i + 4);
  v8bf o;
  o[0] = f2bf(x0.x); o[1] = f2bf(x0.y); o[2] = f2bf(x0.z); o[3] = f2bf(x0.w);
  o[4] = f2bf(x1.x); o[5] = f2bf(x1.y); o[6] = f2bf(x1.z); o[7] = f2bf(x1.w);
  *(v8bf*)(out + i) = o;
}

// ---------------------------------------------------------------------------
// Projection GEMM  C = X @ W^T  (bf16 in, bf16 out into head layouts)
// Wave tile 32x64 (8 WMMAs / K-step), block tile 64x128 (4 waves).
// MODE 0: out[((b*H + h)*S + s)*64 + d]   (Q, K)
// MODE 1: out[((b*H + h)*64 + d)*S + s]   (V transposed for P@V B-frags)
// ---------------------------------------------------------------------------
template <int KDIM, int S, int MODE>
__global__ __launch_bounds__(128)
void proj_gemm_kernel(const bf16_t* __restrict__ X, const bf16_t* __restrict__ W,
                      bf16_t* __restrict__ out) {
  const int lane = threadIdx.x & 31;
  const int wave = threadIdx.x >> 5;
  const int ln = lane & 15, hi = lane >> 4;
  const int m0 = blockIdx.x * 64 + (wave & 1) * 32;
  const int n0 = blockIdx.y * 128 + (wave >> 1) * 64;

  v8f acc[2][4] = {};
  for (int k0 = 0; k0 < KDIM; k0 += 32) {
    if (k0 + 32 < KDIM) {
      __builtin_prefetch(X + (long)(m0 + ln) * KDIM + k0 + 32, 0, 1);
      __builtin_prefetch(W + (long)(n0 + ln) * KDIM + k0 + 32, 0, 1);
    }
    v16bf a0 = load_a16(X, KDIM, m0 + ln, hi, k0);
    v16bf a1 = load_a16(X, KDIM, m0 + 16 + ln, hi, k0);
#pragma unroll
    for (int j = 0; j < 4; ++j) {
      v16bf bj = load_b16(W, KDIM, n0 + j * 16 + ln, hi, k0);
      acc[0][j] = wmma_bf16(a0, bj, acc[0][j]);
      acc[1][j] = wmma_bf16(a1, bj, acc[1][j]);
    }
  }
  // C/D layout: lane holds column n=ln, rows r + 8*hi
#pragma unroll
  for (int i = 0; i < 2; ++i)
#pragma unroll
    for (int j = 0; j < 4; ++j)
#pragma unroll
      for (int r = 0; r < 8; ++r) {
        int m = m0 + i * 16 + r + 8 * hi;
        int n = n0 + j * 16 + ln;
        int b = m / S, s = m % S;      // S is a power-of-two constant
        int h = n >> 6, d = n & 63;
        long idx;
        if (MODE == 0) idx = ((long)(b * HEADS + h) * S + s) * DIM_HEAD + d;
        else           idx = ((long)(b * HEADS + h) * DIM_HEAD + d) * S + s;
        out[idx] = f2bf(acc[i][j][r]);
      }
}

// ---------------------------------------------------------------------------
// Fused flash attention. Block = 4 waves sharing one (b,h); each wave owns 16
// queries. K/V 64x64 tiles are cooperatively staged into LDS once per block
// (4x global-traffic reduction; fragments become ds_load_b128), with the
// split workgroup barrier between stage and compute phases. Online softmax in
// registers; P round-trips through per-wave LDS to convert the score C-layout
// into the A-fragment layout for the P@V WMMAs.
// ---------------------------------------------------------------------------
__global__ __launch_bounds__(128)
void attn_kernel(const bf16_t* __restrict__ Qb, const bf16_t* __restrict__ Kb,
                 const bf16_t* __restrict__ Vt, bf16_t* __restrict__ attn) {
  __shared__ bf16_t Ks[64 * 64];          // [key][d]  8KB
  __shared__ bf16_t Vs[64 * 64];          // [d][key]  8KB
  __shared__ bf16_t pshare[4][16 * 64];   // per-wave P tile, 2KB each
  const int tid = threadIdx.x;
  const int lane = tid & 31;
  const int wave = tid >> 5;
  const int ln = lane & 15, hi = lane >> 4;
  const int bh = blockIdx.x;              // b*H + h
  const int b = bh >> 3, h = bh & 7;
  const int q0 = blockIdx.y * 64 + wave * 16;

  const bf16_t* Qhead = Qb + (long)bh * SQ * DIM_HEAD;
  const bf16_t* Khead = Kb + (long)bh * SKV * DIM_HEAD;
  const bf16_t* Vhead = Vt + (long)bh * DIM_HEAD * SKV;

  // Q tile 16x64 as two persistent A-fragments
  v16bf qa0 = load_a16(Qhead, DIM_HEAD, q0 + ln, hi, 0);
  v16bf qa1 = load_a16(Qhead, DIM_HEAD, q0 + ln, hi, 32);

  float mrow[8], lrow[8];
#pragma unroll
  for (int r = 0; r < 8; ++r) { mrow[r] = -1e30f; lrow[r] = 0.f; }
  v8f oacc[4] = {};

  const float scale = 0.125f;  // DIM_HEAD^-0.5

  for (int kv0 = 0; kv0 < SKV; kv0 += 64) {
    __syncthreads();  // previous tile fully consumed
    // ---- cooperative stage: K tile (row-contiguous) ----
#pragma unroll
    for (int i = 0; i < 4; ++i) {
      int c = tid + 128 * i;              // 512 chunks of 16B
      ((v8bf*)Ks)[c] = *(const v8bf*)(Khead + (long)kv0 * DIM_HEAD + c * 8);
    }
    // ---- cooperative stage: V^T tile (row d, 64 keys) ----
#pragma unroll
    for (int i = 0; i < 4; ++i) {
      int c = tid + 128 * i;
      int d = c >> 3, off = (c & 7) * 8;
      ((v8bf*)Vs)[c] = *(const v8bf*)(Vhead + (long)d * SKV + kv0 + off);
    }
    __syncthreads();  // tile visible to all waves

    // scores S = scale * Q @ K^T  (4 n-tiles x 2 K-steps = 8 WMMAs)
    v8f s[4];
#pragma unroll
    for (int nt = 0; nt < 4; ++nt) {
      v8f sa = {};
      sa = wmma_bf16(qa0, load_b16(Ks, DIM_HEAD, nt * 16 + ln, hi, 0), sa);
      sa = wmma_bf16(qa1, load_b16(Ks, DIM_HEAD, nt * 16 + ln, hi, 32), sa);
#pragma unroll
      for (int r = 0; r < 8; ++r) sa[r] *= scale;
      s[nt] = sa;
    }
    // online softmax; a lane's rows are r + 8*hi, reduce across 16-lane half
    float rsum[8], alpha[8];
#pragma unroll
    for (int r = 0; r < 8; ++r) {
      float m = fmaxf(fmaxf(s[0][r], s[1][r]), fmaxf(s[2][r], s[3][r]));
#pragma unroll
      for (int off = 1; off < 16; off <<= 1)
        m = fmaxf(m, __shfl_xor(m, off, 16));
      float mn = fmaxf(mrow[r], m);
      alpha[r] = __expf(mrow[r] - mn);
      mrow[r] = mn;
      rsum[r] = 0.f;
    }
#pragma unroll
    for (int nt = 0; nt < 4; ++nt)
#pragma unroll
      for (int r = 0; r < 8; ++r) {
        float p = __expf(s[nt][r] - mrow[r]);
        s[nt][r] = p;
        rsum[r] += p;
      }
#pragma unroll
    for (int r = 0; r < 8; ++r) {
      float t = rsum[r];
#pragma unroll
      for (int off = 1; off < 16; off <<= 1) t += __shfl_xor(t, off, 16);
      lrow[r] = lrow[r] * alpha[r] + t;
    }
    // P: C-layout -> LDS row-major -> A-fragments (same-wave, DScnt-ordered)
    bf16_t* P = pshare[wave];
#pragma unroll
    for (int nt = 0; nt < 4; ++nt)
#pragma unroll
      for (int r = 0; r < 8; ++r)
        P[(r + 8 * hi) * 64 + nt * 16 + ln] = f2bf(s[nt][r]);
    v16bf pa0 = load_a16(P, 64, ln, hi, 0);
    v16bf pa1 = load_a16(P, 64, ln, hi, 32);
    // rescale running output, then accumulate P @ V (8 WMMAs)
#pragma unroll
    for (int dt = 0; dt < 4; ++dt)
#pragma unroll
      for (int r = 0; r < 8; ++r) oacc[dt][r] *= alpha[r];
#pragma unroll
    for (int dt = 0; dt < 4; ++dt) {
      oacc[dt] = wmma_bf16(pa0, load_b16(Vs, 64, dt * 16 + ln, hi, 0), oacc[dt]);
      oacc[dt] = wmma_bf16(pa1, load_b16(Vs, 64, dt * 16 + ln, hi, 32), oacc[dt]);
    }
  }
  // normalize (v_rcp_f32) + store as [B, Sq, H*D] bf16
  float linv[8];
#pragma unroll
  for (int r = 0; r < 8; ++r) linv[r] = __builtin_amdgcn_rcpf(lrow[r]);
#pragma unroll
  for (int dt = 0; dt < 4; ++dt)
#pragma unroll
    for (int r = 0; r < 8; ++r) {
      int q = q0 + r + 8 * hi;
      int d = dt * 16 + ln;
      attn[(long)(b * SQ + q) * INNER + h * DIM_HEAD + d] =
          f2bf(oacc[dt][r] * linv[r]);
    }
}

// ---------------------------------------------------------------------------
// Out projection  Y = attn @ Wo^T + bo  (bf16 in, fp32 out), 32x64 wave tile
// ---------------------------------------------------------------------------
__global__ __launch_bounds__(128)
void outproj_kernel(const bf16_t* __restrict__ A, const bf16_t* __restrict__ Wo,
                    const float* __restrict__ bo, float* __restrict__ Y) {
  const int lane = threadIdx.x & 31;
  const int wave = threadIdx.x >> 5;
  const int ln = lane & 15, hi = lane >> 4;
  const int m0 = blockIdx.x * 64 + (wave & 1) * 32;
  const int n0 = blockIdx.y * 128 + (wave >> 1) * 64;

  v8f acc[2][4] = {};
  for (int k0 = 0; k0 < INNER; k0 += 32) {
    if (k0 + 32 < INNER) {
      __builtin_prefetch(A + (long)(m0 + ln) * INNER + k0 + 32, 0, 1);
      __builtin_prefetch(Wo + (long)(n0 + ln) * INNER + k0 + 32, 0, 1);
    }
    v16bf a0 = load_a16(A, INNER, m0 + ln, hi, k0);
    v16bf a1 = load_a16(A, INNER, m0 + 16 + ln, hi, k0);
#pragma unroll
    for (int j = 0; j < 4; ++j) {
      v16bf bj = load_b16(Wo, INNER, n0 + j * 16 + ln, hi, k0);
      acc[0][j] = wmma_bf16(a0, bj, acc[0][j]);
      acc[1][j] = wmma_bf16(a1, bj, acc[1][j]);
    }
  }
#pragma unroll
  for (int i = 0; i < 2; ++i)
#pragma unroll
    for (int j = 0; j < 4; ++j)
#pragma unroll
      for (int r = 0; r < 8; ++r) {
        int m = m0 + i * 16 + r + 8 * hi;
        int n = n0 + j * 16 + ln;
        Y[(long)m * QUERY_DIM + n] = acc[i][j][r] + bo[n];
      }
}

extern "C" void kernel_launch(void* const* d_in, const int* in_sizes, int n_in,
                              void* d_out, int out_size, void* d_ws, size_t ws_size,
                              hipStream_t stream) {
  const float* hs  = (const float*)d_in[0];  // [B, Sq, 512]
  const float* enc = (const float*)d_in[1];  // [B, Skv, 768]
  const float* Wq  = (const float*)d_in[2];  // [512, 512]
  const float* Wk  = (const float*)d_in[3];  // [512, 768]
  const float* Wv  = (const float*)d_in[4];  // [512, 768]
  const float* Wo  = (const float*)d_in[5];  // [512, 512]
  const float* bo  = (const float*)d_in[6];  // [512]
  float* out = (float*)d_out;

  // workspace (bf16): head tensors + bf16 copies of activations/weights (~34MB)
  bf16_t* ws = (bf16_t*)d_ws;
  const long QN  = (long)BATCH * HEADS * SQ * DIM_HEAD;   // 4M
  const long KN  = (long)BATCH * HEADS * SKV * DIM_HEAD;  // 1M
  const long HSN = (long)BATCH * SQ * QUERY_DIM;          // 4M
  const long ENN = (long)BATCH * SKV * CROSS_DIM;         // 1.5M
  const long WQN = (long)INNER * QUERY_DIM;               // 256K
  const long WKN = (long)INNER * CROSS_DIM;               // 384K
  bf16_t* Qb   = ws;
  bf16_t* Kb   = Qb + QN;
  bf16_t* Vt   = Kb + KN;
  bf16_t* attn = Vt + KN;
  bf16_t* hsb  = attn + QN;
  bf16_t* encb = hsb + HSN;
  bf16_t* Wqb  = encb + ENN;
  bf16_t* Wkb  = Wqb + WQN;
  bf16_t* Wvb  = Wkb + WKN;
  bf16_t* Wob  = Wvb + WKN;

  // one-shot fp32 -> bf16 conversions (memory bound)
  auto cvt = [&](const float* src, bf16_t* dst, long n) {
    cvt_bf16_kernel<<<dim3((unsigned)((n / 8 + 255) / 256)), 256, 0, stream>>>(
        src, dst, n);
  };
  cvt(hs,  hsb,  HSN);
  cvt(enc, encb, ENN);
  cvt(Wq,  Wqb,  WQN);
  cvt(Wk,  Wkb,  WKN);
  cvt(Wv,  Wvb,  WKN);
  cvt(Wo,  Wob,  WQN);

  dim3 blk(128);
  proj_gemm_kernel<QUERY_DIM, SQ, 0>
      <<<dim3((BATCH * SQ) / 64, INNER / 128), blk, 0, stream>>>(hsb, Wqb, Qb);
  proj_gemm_kernel<CROSS_DIM, SKV, 0>
      <<<dim3((BATCH * SKV) / 64, INNER / 128), blk, 0, stream>>>(encb, Wkb, Kb);
  proj_gemm_kernel<CROSS_DIM, SKV, 1>
      <<<dim3((BATCH * SKV) / 64, INNER / 128), blk, 0, stream>>>(encb, Wvb, Vt);
  attn_kernel<<<dim3(BATCH * HEADS, SQ / 64), blk, 0, stream>>>(Qb, Kb, Vt, attn);
  outproj_kernel<<<dim3((BATCH * SQ) / 64, QUERY_DIM / 128), blk, 0, stream>>>(
      attn, Wob, bo, out);
}
